// WindowAttention_26929444946408
// MI455X (gfx1250) — compile-verified, matrix operand order checked by reading, not run on hardware
//
#include <hip/hip_runtime.h>

// ---------------------------------------------------------------------------
// Swin window attention for gfx1250 (MI455X), bf16 WMMA pipeline.
//   x[2048,64,512] -> qkv gemm -> per-head attention -> proj gemm -> out
// GEMMs are M-tiled only: each block stages its 64x512 activation tile in LDS
// once (x does not fit in the 192MB L2, so N-blocking would multiply HBM
// traffic), then sweeps all output columns against L2-resident bf16 weights.
// ---------------------------------------------------------------------------

typedef __attribute__((ext_vector_type(16))) __bf16 v16bf_t;
typedef __attribute__((ext_vector_type(8)))  __bf16 v8bf_t;
typedef __attribute__((ext_vector_type(8)))  float  v8f_t;

#define B_WIN   2048
#define NW_MASK 256
#define HEADS   16
#define NTOK    64
#define CDIM    512
#define DH      32   // head dim

__device__ __forceinline__ v8f_t wmma_bf16(v16bf_t a, v16bf_t b, v8f_t c) {
  return __builtin_amdgcn_wmma_f32_16x16x32_bf16(
      /*neg_a=*/false, a, /*neg_b=*/false, b,
      /*c_mod=*/(short)0, c, /*reuse_a=*/false, /*reuse_b=*/false);
}

// 16x32 bf16 tile in WMMA A/B striping. lane<16: row=lane, K {0..7,16..23};
// lane>=16: row=lane-16, K {8..15,24..31}. Rows must be 16B aligned.
__device__ __forceinline__ v16bf_t load_tile_bf16(const __bf16* base, int ld) {
  int lane = threadIdx.x & 31;
  int row  = lane & 15;
  int kb   = (lane & 16) ? 8 : 0;
  const __bf16* p = base + row * ld + kb;
  v8bf_t lo = *reinterpret_cast<const v8bf_t*>(p);
  v8bf_t hi = *reinterpret_cast<const v8bf_t*>(p + 16);
  return __builtin_shufflevector(lo, hi,
      0,1,2,3,4,5,6,7,8,9,10,11,12,13,14,15);
}

// ---------------------------------------------------------------------------
// Prep kernels
// ---------------------------------------------------------------------------
__global__ void cvt_f32_bf16(const float* __restrict__ s,
                             __bf16* __restrict__ d, int n) {
  for (int i = blockIdx.x * blockDim.x + threadIdx.x; i < n;
       i += gridDim.x * blockDim.x)
    d[i] = (__bf16)s[i];
}

// rpb_g[h][i*64+j] = rpb_table[rel_index[i*64+j]][h]
__global__ void gather_rpb(const float* __restrict__ tab,
                           const int* __restrict__ idx,
                           float* __restrict__ rpb_g) {
  int t = blockIdx.x * blockDim.x + threadIdx.x;   // 0..65535
  int h  = t >> 12;
  int ij = t & 4095;
  rpb_g[t] = tab[idx[ij] * HEADS + h];
}

// ---------------------------------------------------------------------------
// QKV GEMM: qkv = x @ qkv_w^T (+bias), scatter to q/k/v per-head bf16 buffers.
// grid 2048, block 256 (8 waves). Block stages x tile [64x512] bf16 in LDS.
// Wave (strip = wv&3, half = wv>>2) sweeps 12 chunks of 64 output columns.
// ---------------------------------------------------------------------------
__global__ __launch_bounds__(256)
void qkv_kernel(const float* __restrict__ x, const __bf16* __restrict__ wq,
                const float* __restrict__ qb, const float* __restrict__ vb,
                __bf16* __restrict__ q_s, __bf16* __restrict__ k_s,
                __bf16* __restrict__ v_t) {
  __shared__ __bf16 Xt[64 * CDIM];          // 64 KB
  int m0  = blockIdx.x * 64;
  int tid = threadIdx.x;

  // Cooperative fp32 load -> bf16 convert -> LDS (linear copy, float4 wide).
  const float* xb = x + (size_t)m0 * CDIM;
  for (int i = tid * 4; i < 64 * CDIM; i += 256 * 4) {
    float4 f = *reinterpret_cast<const float4*>(xb + i);
    Xt[i + 0] = (__bf16)f.x;
    Xt[i + 1] = (__bf16)f.y;
    Xt[i + 2] = (__bf16)f.z;
    Xt[i + 3] = (__bf16)f.w;
  }
  __syncthreads();

  int wv    = tid >> 5;
  int lane  = tid & 31;
  int strip = wv & 3;                        // 16-row strip within tile
  int half  = wv >> 1 >> 1;                  // 0/1: which 768-col half
  const __bf16* arow = Xt + strip * 16 * CDIM;
  int nloc = lane & 15;
  int ms   = lane >> 4;
  const float scale = 0.17677669529663687f;  // 32^-0.5

  for (int c = 0; c < 12; ++c) {
    int n0 = (half * 12 + c) * 64;           // 0..1472
    v8f_t acc[4] = {};
    for (int k0 = 0; k0 < CDIM; k0 += 32) {
      v16bf_t a = load_tile_bf16(arow + k0, CDIM);   // ds_load from LDS
#pragma unroll
      for (int nt = 0; nt < 4; ++nt) {
        v16bf_t b =
            load_tile_bf16(wq + (size_t)(n0 + nt * 16) * CDIM + k0, CDIM);
        acc[nt] = wmma_bf16(a, b, acc[nt]);
      }
    }
    // Epilogue: add bias, scale q, scatter to per-head q/k/v (v transposed).
#pragma unroll
    for (int nt = 0; nt < 4; ++nt) {
      int cg  = n0 + nt * 16 + nloc;         // 0..1535
      int sec = cg >> 9;                     // 0=q 1=k 2=v
      int cc  = cg & 511;
      int head = cc >> 5, d = cc & 31;
      float bias = (sec == 0) ? qb[cc] : ((sec == 2) ? vb[cc] : 0.0f);
#pragma unroll
      for (int r = 0; r < 8; ++r) {
        int mg  = m0 + strip * 16 + r + ms * 8;
        int bw  = mg >> 6;
        int tok = mg & 63;
        float val = acc[nt][r] + bias;
        size_t hb = (size_t)bw * HEADS + head;
        if (sec == 0)
          q_s[(hb * NTOK + tok) * DH + d] = (__bf16)(val * scale);
        else if (sec == 1)
          k_s[(hb * NTOK + tok) * DH + d] = (__bf16)val;
        else
          v_t[(hb * DH + d) * NTOK + tok] = (__bf16)val;  // transposed
      }
    }
  }
}

// ---------------------------------------------------------------------------
// Attention: per (window b, head h). grid (2048,16), block 128.
// Wave w handles rows [16w,16w+16): S=qk^T (+rpb+mask), softmax, P@V.
// ---------------------------------------------------------------------------
__global__ __launch_bounds__(128)
void attn_kernel(const __bf16* __restrict__ q_s, const __bf16* __restrict__ k_s,
                 const __bf16* __restrict__ v_t, const float* __restrict__ rpb_g,
                 const float* __restrict__ mask, __bf16* __restrict__ attn_out) {
  int b = blockIdx.x, h = blockIdx.y;
  int wv   = threadIdx.x >> 5;
  int lane = threadIdx.x & 31;
  int nloc = lane & 15, ms = lane >> 4;

  __shared__ __bf16 Pst[4 * 16 * NTOK];      // 8 KB: per-wave P strips

  const __bf16* qh = q_s + (size_t)(b * HEADS + h) * NTOK * DH;
  const __bf16* kh = k_s + (size_t)(b * HEADS + h) * NTOK * DH;
  const __bf16* vh = v_t + (size_t)(b * HEADS + h) * DH * NTOK;

  // ---- S = (q*scale) @ k^T : 4 tiles of 16x16, K=32 (one WMMA step) ----
  v8f_t s[4] = {};
  {
    v16bf_t a = load_tile_bf16(qh + (size_t)(wv * 16) * DH, DH);
#pragma unroll
    for (int nt = 0; nt < 4; ++nt) {
      v16bf_t kb = load_tile_bf16(kh + (size_t)(nt * 16) * DH, DH);
      s[nt] = wmma_bf16(a, kb, s[nt]);
    }
  }

  // ---- add relative-position bias + window mask (C-layout indexing) ----
  const float* mrow = mask  + (size_t)(b & (NW_MASK - 1)) * NTOK * NTOK;
  const float* rrow = rpb_g + (size_t)h * NTOK * NTOK;
#pragma unroll
  for (int nt = 0; nt < 4; ++nt) {
    int col = nt * 16 + nloc;
#pragma unroll
    for (int r = 0; r < 8; ++r) {
      int row = wv * 16 + r + ms * 8;
      s[nt][r] += rrow[row * NTOK + col] + mrow[row * NTOK + col];
    }
  }

  // ---- softmax over each row (row lives in one 16-lane half) ----
  __bf16* strip = &Pst[wv * 16 * NTOK];
#pragma unroll
  for (int r = 0; r < 8; ++r) {
    float m = fmaxf(fmaxf(s[0][r], s[1][r]), fmaxf(s[2][r], s[3][r]));
    m = fmaxf(m, __shfl_xor(m, 1, 16));
    m = fmaxf(m, __shfl_xor(m, 2, 16));
    m = fmaxf(m, __shfl_xor(m, 4, 16));
    m = fmaxf(m, __shfl_xor(m, 8, 16));
    float e0 = __expf(s[0][r] - m);
    float e1 = __expf(s[1][r] - m);
    float e2 = __expf(s[2][r] - m);
    float e3 = __expf(s[3][r] - m);
    float sum = e0 + e1 + e2 + e3;
    sum += __shfl_xor(sum, 1, 16);
    sum += __shfl_xor(sum, 2, 16);
    sum += __shfl_xor(sum, 4, 16);
    sum += __shfl_xor(sum, 8, 16);
    float inv = 1.0f / sum;
    int rowl = r + ms * 8;
    strip[rowl * NTOK + 0 * 16 + nloc] = (__bf16)(e0 * inv);
    strip[rowl * NTOK + 1 * 16 + nloc] = (__bf16)(e1 * inv);
    strip[rowl * NTOK + 2 * 16 + nloc] = (__bf16)(e2 * inv);
    strip[rowl * NTOK + 3 * 16 + nloc] = (__bf16)(e3 * inv);
  }
  __syncthreads();

  // ---- out = P @ V : A from LDS strip, B from transposed V ----
  v8f_t o[2] = {};
#pragma unroll
  for (int kk = 0; kk < NTOK; kk += 32) {
    v16bf_t pa = load_tile_bf16(strip + kk, NTOK);
#pragma unroll
    for (int nt = 0; nt < 2; ++nt) {
      v16bf_t vb = load_tile_bf16(vh + (size_t)(nt * 16) * NTOK + kk, NTOK);
      o[nt] = wmma_bf16(pa, vb, o[nt]);
    }
  }

  __bf16* ob = attn_out + ((size_t)b * NTOK) * CDIM + h * DH;
#pragma unroll
  for (int nt = 0; nt < 2; ++nt) {
#pragma unroll
    for (int r = 0; r < 8; ++r) {
      int tok = wv * 16 + r + ms * 8;
      int d   = nt * 16 + nloc;
      ob[(size_t)tok * CDIM + d] = (__bf16)o[nt][r];
    }
  }
}

// ---------------------------------------------------------------------------
// Proj GEMM: out = attn_bf @ proj_w^T + proj_b (fp32 out).
// grid 2048, block 256 (8 waves). Block stages A tile [64x512] bf16 in LDS.
// Wave (strip = wv&3, half = wv>>2) sweeps 4 chunks of 64 output columns.
// ---------------------------------------------------------------------------
__global__ __launch_bounds__(256)
void proj_kernel(const __bf16* __restrict__ a_in, const __bf16* __restrict__ w,
                 const float* __restrict__ pb, float* __restrict__ out) {
  __shared__ __bf16 At[64 * CDIM];           // 64 KB
  int m0  = blockIdx.x * 64;
  int tid = threadIdx.x;

  const __bf16* ab = a_in + (size_t)m0 * CDIM;
  for (int i = tid * 8; i < 64 * CDIM; i += 256 * 8)
    *reinterpret_cast<v8bf_t*>(&At[i]) =
        *reinterpret_cast<const v8bf_t*>(ab + i);
  __syncthreads();

  int wv    = tid >> 5;
  int lane  = tid & 31;
  int strip = wv & 3;
  int half  = wv >> 1 >> 1;                  // 0/1: which 256-col half
  const __bf16* arow = At + strip * 16 * CDIM;
  int nloc = lane & 15, ms = lane >> 4;

  for (int c = 0; c < 4; ++c) {
    int n0 = (half * 4 + c) * 64;            // 0..448
    v8f_t acc[4] = {};
    for (int k0 = 0; k0 < CDIM; k0 += 32) {
      v16bf_t a = load_tile_bf16(arow + k0, CDIM);
#pragma unroll
      for (int nt = 0; nt < 4; ++nt) {
        v16bf_t b =
            load_tile_bf16(w + (size_t)(n0 + nt * 16) * CDIM + k0, CDIM);
        acc[nt] = wmma_bf16(a, b, acc[nt]);
      }
    }
#pragma unroll
    for (int nt = 0; nt < 4; ++nt) {
      int cg = n0 + nt * 16 + nloc;
      float bias = pb[cg];
#pragma unroll
      for (int r = 0; r < 8; ++r) {
        int mg = m0 + strip * 16 + r + ms * 8;
        out[(size_t)mg * CDIM + cg] = acc[nt][r] + bias;
      }
    }
  }
}

// ---------------------------------------------------------------------------
// Launch
// ---------------------------------------------------------------------------
extern "C" void kernel_launch(void* const* d_in, const int* in_sizes, int n_in,
                              void* d_out, int out_size, void* d_ws, size_t ws_size,
                              hipStream_t stream) {
  const float* x         = (const float*)d_in[0];
  const float* mask      = (const float*)d_in[1];
  const float* qkv_w     = (const float*)d_in[2];
  const float* q_bias    = (const float*)d_in[3];
  const float* v_bias    = (const float*)d_in[4];
  const float* rpb_table = (const float*)d_in[5];
  const float* proj_w    = (const float*)d_in[6];
  const float* proj_b    = (const float*)d_in[7];
  const int*   rel_index = (const int*)d_in[8];
  float* out = (float*)d_out;

  char* ws = (char*)d_ws;
  size_t off = 0;
  auto take = [&](size_t bytes) -> void* {
    void* p = ws + off;
    off += (bytes + 255) & ~(size_t)255;
    return p;
  };
  __bf16* qkvw_bf  = (__bf16*)take((size_t)3 * CDIM * CDIM * 2);
  __bf16* projw_bf = (__bf16*)take((size_t)CDIM * CDIM * 2);
  float*  rpb_g    = (float*) take((size_t)HEADS * NTOK * NTOK * 4);
  __bf16* q_s      = (__bf16*)take((size_t)B_WIN * HEADS * NTOK * DH * 2);
  __bf16* k_s      = (__bf16*)take((size_t)B_WIN * HEADS * NTOK * DH * 2);
  __bf16* v_t      = (__bf16*)take((size_t)B_WIN * HEADS * DH * NTOK * 2);
  __bf16* attn_bf  = (__bf16*)take((size_t)B_WIN * NTOK * CDIM * 2);

  cvt_f32_bf16<<<3072, 256, 0, stream>>>(qkv_w, qkvw_bf, 3 * CDIM * CDIM);
  cvt_f32_bf16<<<1024, 256, 0, stream>>>(proj_w, projw_bf, CDIM * CDIM);
  gather_rpb<<<256, 256, 0, stream>>>(rpb_table, rel_index, rpb_g);

  qkv_kernel<<<B_WIN, 256, 0, stream>>>(x, qkvw_bf, q_bias, v_bias,
                                        q_s, k_s, v_t);
  attn_kernel<<<dim3(B_WIN, HEADS), 128, 0, stream>>>(q_s, k_s, v_t, rpb_g,
                                                      mask, attn_bf);
  proj_kernel<<<B_WIN, 256, 0, stream>>>(attn_bf, projw_bf, proj_b, out);
}
